// Attention_11304353923345
// MI455X (gfx1250) — compile-verified, matrix operand order checked by reading, not run on hardware
//
#include <hip/hip_runtime.h>
#include <hip/hip_bf16.h>
#include <stdint.h>

typedef __attribute__((ext_vector_type(16))) __bf16 v16bf;
typedef __attribute__((ext_vector_type(8)))  __bf16 v8bf;
typedef __attribute__((ext_vector_type(8)))  float  v8f;

// ---------- helpers ----------

static __device__ __forceinline__ __bf16 f2bf(float f) {
  uint32_t u = __builtin_bit_cast(uint32_t, f);
  uint32_t r = u + 0x7FFFu + ((u >> 16) & 1u);   // round-to-nearest-even
  uint16_t h = (uint16_t)(r >> 16);
  return __builtin_bit_cast(__bf16, h);
}

// Load a 16x32 bf16 WMMA A-fragment (or a 32x16 B-fragment stored as
// [n][k] row-major, i.e. B^T) with row stride `ld` elements.
// Works for both global and LDS pointers (address space inferred after inline).
static __device__ __forceinline__ v16bf load_frag16(const __bf16* base, int ld) {
  const int lane = threadIdx.x & 31;
  const int row  = lane & 15;
  const int kb   = (lane >> 4) << 3;     // 0 or 8
  const __bf16* p = base + row * ld + kb;
  v8bf lo = *(const v8bf*)(p);           // K = kb .. kb+7
  v8bf hi = *(const v8bf*)(p + 16);      // K = kb+16 .. kb+23
  return __builtin_shufflevector(lo, hi, 0,1,2,3,4,5,6,7,8,9,10,11,12,13,14,15);
}

static __device__ __forceinline__ v8f wmma_bf16(v16bf a, v16bf b, v8f c) {
  return __builtin_amdgcn_wmma_f32_16x16x32_bf16(false, a, false, b, (short)0, c,
                                                 false, false);
}

// 16-byte global->LDS copy via the CDNA5 async DMA path (ASYNCcnt-tracked).
// LDS address = low 32 bits of the flat pointer (aperture mapping, ISA 10.2).
static __device__ __forceinline__ void cp16_g2l(const __bf16* g, __bf16* l) {
#if defined(__HIP_DEVICE_COMPILE__)
  asm volatile("global_load_async_to_lds_b128 %0, %1, off"
               :
               : "v"((uint32_t)(uintptr_t)l), "v"(g)
               : "memory");
#else
  *(v8bf*)l = *(const v8bf*)g;   // host-pass placeholder (never executed)
#endif
}

static __device__ __forceinline__ void wait_async_lds() {
#if defined(__HIP_DEVICE_COMPILE__)
  asm volatile("s_wait_asynccnt 0" ::: "memory");
#endif
}

// ---------- kernel 1: f32 -> bf16 convert ----------

__global__ void cvt_f32_bf16(const float* __restrict__ src,
                             __bf16* __restrict__ dst, int n) {
  int i = blockIdx.x * blockDim.x + threadIdx.x;
  if (i < n) dst[i] = f2bf(src[i]);
}

// ---------- kernel 2: fused QKV GEMM + bias + scale + scatter ----------
__global__ __launch_bounds__(256) void qkv_gemm(
    const __bf16* __restrict__ xb, const __bf16* __restrict__ wq,
    const float* __restrict__ qkv_b,
    __bf16* __restrict__ Qs, __bf16* __restrict__ Ks, __bf16* __restrict__ Vt) {
  const int N = 4096, C = 1024;
  const int wid  = threadIdx.x >> 5;
  const int lane = threadIdx.x & 31;
  const int m0 = blockIdx.x * 64  + (wid & 1) * 32;   // 2 M-waves
  const int n0 = blockIdx.y * 256 + (wid >> 1) * 64;  // 4 N-waves

  v8f c[2][4];
#pragma unroll
  for (int i = 0; i < 2; ++i)
#pragma unroll
    for (int f = 0; f < 4; ++f) { v8f z = {}; c[i][f] = z; }

  for (int kk = 0; kk < C; kk += 32) {
    // warm L2 two K-steps ahead (global_prefetch_b8)
    if (kk + 64 < C) {
      __builtin_prefetch(xb + (size_t)m0 * C + kk + 64, 0, 1);
      __builtin_prefetch(wq + (size_t)n0 * C + kk + 64, 0, 1);
    }
    v16bf a0 = load_frag16(xb + (size_t)m0 * C + kk, C);
    v16bf a1 = load_frag16(xb + (size_t)(m0 + 16) * C + kk, C);
    v16bf b0 = load_frag16(wq + (size_t)(n0 +  0) * C + kk, C);
    v16bf b1 = load_frag16(wq + (size_t)(n0 + 16) * C + kk, C);
    v16bf b2 = load_frag16(wq + (size_t)(n0 + 32) * C + kk, C);
    v16bf b3 = load_frag16(wq + (size_t)(n0 + 48) * C + kk, C);
    c[0][0] = wmma_bf16(a0, b0, c[0][0]);
    c[0][1] = wmma_bf16(a0, b1, c[0][1]);
    c[0][2] = wmma_bf16(a0, b2, c[0][2]);
    c[0][3] = wmma_bf16(a0, b3, c[0][3]);
    c[1][0] = wmma_bf16(a1, b0, c[1][0]);
    c[1][1] = wmma_bf16(a1, b1, c[1][1]);
    c[1][2] = wmma_bf16(a1, b2, c[1][2]);
    c[1][3] = wmma_bf16(a1, b3, c[1][3]);
  }

  const int cl = lane & 15;
  const int mb = (lane >> 4) << 3;
  const float scale = 0.125f;        // 64^-0.5, folded into Q
#pragma unroll
  for (int i = 0; i < 2; ++i)
#pragma unroll
    for (int f = 0; f < 4; ++f) {
      int j = n0 + f * 16 + cl;
      int which = j >> 10;
      int rem = j & 1023;
      int h = rem >> 6, d = rem & 63;
      float bias = qkv_b[j];
#pragma unroll
      for (int r = 0; r < 8; ++r) {
        int m = m0 + i * 16 + mb + r;
        float v = c[i][f][r] + bias;
        if (which == 0)      Qs[((size_t)h * N + m) * 64 + d] = f2bf(v * scale);
        else if (which == 1) Ks[((size_t)h * N + m) * 64 + d] = f2bf(v);
        else                 Vt[((size_t)h * 64 + d) * N + m] = f2bf(v);
      }
    }
}

// ---------- kernel 3: flash attention with async LDS staging ----------
// Block = 8 waves x 16 query rows = 128 rows, one head.
// K tile (32x64) and V^T tile (64x32) are staged in LDS once per block
// (8x VMEM reuse), double-buffered with async global->LDS DMA.
__global__ __launch_bounds__(256) void attn_kernel(
    const __bf16* __restrict__ Qs, const __bf16* __restrict__ Ks,
    const __bf16* __restrict__ Vt, __bf16* __restrict__ A2) {
  const int N = 4096, D = 64, C = 1024;
  __shared__ __attribute__((aligned(16))) __bf16 Ktile[2][32][64];  // [key][d]
  __shared__ __attribute__((aligned(16))) __bf16 Vtile[2][64][32];  // [d][key]
  __shared__ __attribute__((aligned(16))) __bf16 Plds[8][16][40];

  const int tid  = threadIdx.x;
  const int wid  = tid >> 5;
  const int lane = tid & 31;
  const int h  = blockIdx.y;
  const int q0 = blockIdx.x * 128 + wid * 16;

  const __bf16* Qh = Qs + (size_t)h * N * D;
  const __bf16* Kh = Ks + (size_t)h * N * D;
  const __bf16* Vh = Vt + (size_t)h * D * N;

  // Q fragments resident for the whole key loop (scale folded in).
  v16bf a0 = load_frag16(Qh + (size_t)q0 * D +  0, D);
  v16bf a1 = load_frag16(Qh + (size_t)q0 * D + 32, D);

  v8f o[4];
#pragma unroll
  for (int f = 0; f < 4; ++f) { v8f z = {}; o[f] = z; }
  float mi[8], li[8];
#pragma unroll
  for (int r = 0; r < 8; ++r) { mi[r] = -3.0e38f; li[r] = 0.0f; }

  const int cl = lane & 15;
  const int mb = (lane >> 4) << 3;

  // --- stage tile 0 ---
  {
    int krow = tid >> 3, kseg = tid & 7;               // 32 rows x 8 segs
    cp16_g2l(Kh + (size_t)krow * D + kseg * 8, &Ktile[0][krow][kseg * 8]);
    int vrow = tid >> 2, vseg = tid & 3;               // 64 rows x 4 segs
    cp16_g2l(Vh + (size_t)vrow * N + vseg * 8, &Vtile[0][vrow][vseg * 8]);
  }
  wait_async_lds();
  __syncthreads();

  int cur = 0;
  for (int kb = 0; kb < N; kb += 32) {
    // --- prefetch next tile into the other buffer (async DMA) ---
    if (kb + 32 < N) {
      int nb = kb + 32;
      int krow = tid >> 3, kseg = tid & 7;
      cp16_g2l(Kh + (size_t)(nb + krow) * D + kseg * 8,
               &Ktile[cur ^ 1][krow][kseg * 8]);
      int vrow = tid >> 2, vseg = tid & 3;
      cp16_g2l(Vh + (size_t)vrow * N + nb + vseg * 8,
               &Vtile[cur ^ 1][vrow][vseg * 8]);
    }

    // ---- S = Q * K^T from LDS (two 16x16 C-frags) ----
    v8f s0 = {}, s1 = {};
    v16bf bk00 = load_frag16(&Ktile[cur][ 0][ 0], 64);
    v16bf bk01 = load_frag16(&Ktile[cur][ 0][32], 64);
    v16bf bk10 = load_frag16(&Ktile[cur][16][ 0], 64);
    v16bf bk11 = load_frag16(&Ktile[cur][16][32], 64);
    s0 = wmma_bf16(a0, bk00, s0);
    s0 = wmma_bf16(a1, bk01, s0);
    s1 = wmma_bf16(a0, bk10, s1);
    s1 = wmma_bf16(a1, bk11, s1);

    // ---- online softmax ----
    float mnew[8], alpha[8], rs[8];
#pragma unroll
    for (int r = 0; r < 8; ++r) {
      float v = fmaxf(s0[r], s1[r]);
#pragma unroll
      for (int off = 1; off < 16; off <<= 1)
        v = fmaxf(v, __shfl_xor(v, off, 32));
      mnew[r]  = fmaxf(mi[r], v);
      alpha[r] = __expf(mi[r] - mnew[r]);
      mi[r]    = mnew[r];
    }
#pragma unroll
    for (int r = 0; r < 8; ++r) {
      float p0 = __expf(s0[r] - mnew[r]);
      float p1 = __expf(s1[r] - mnew[r]);
      s0[r] = p0; s1[r] = p1;
      float t = p0 + p1;
#pragma unroll
      for (int off = 1; off < 16; off <<= 1)
        t += __shfl_xor(t, off, 32);
      rs[r] = t;
    }
#pragma unroll
    for (int r = 0; r < 8; ++r) li[r] = li[r] * alpha[r] + rs[r];
#pragma unroll
    for (int f = 0; f < 4; ++f)
#pragma unroll
      for (int r = 0; r < 8; ++r) o[f][r] *= alpha[r];

    // ---- relayout P: C-frag layout -> A-frag layout via per-wave LDS ----
#pragma unroll
    for (int r = 0; r < 8; ++r) {
      Plds[wid][mb + r][cl]      = f2bf(s0[r]);
      Plds[wid][mb + r][16 + cl] = f2bf(s1[r]);
    }
    asm volatile("s_wait_dscnt 0" ::: "memory");
    v16bf pa;
    {
      int row = lane & 15;
      int kk  = (lane >> 4) << 3;
      v8bf lo = *(const v8bf*)&Plds[wid][row][kk];
      v8bf hi = *(const v8bf*)&Plds[wid][row][kk + 16];
      pa = __builtin_shufflevector(lo, hi, 0,1,2,3,4,5,6,7,8,9,10,11,12,13,14,15);
    }

    // ---- O += P * V from LDS ----
#pragma unroll
    for (int f = 0; f < 4; ++f) {
      v16bf bv = load_frag16(&Vtile[cur][f * 16][0], 32);
      o[f] = wmma_bf16(pa, bv, o[f]);
    }

    // next tile ready + all waves done with current buffers
    wait_async_lds();
    __syncthreads();
    cur ^= 1;
  }

  // ---- normalize and store ----
#pragma unroll
  for (int r = 0; r < 8; ++r) {
    float inv = 1.0f / li[r];
#pragma unroll
    for (int f = 0; f < 4; ++f) o[f][r] *= inv;
  }
#pragma unroll
  for (int f = 0; f < 4; ++f)
#pragma unroll
    for (int r = 0; r < 8; ++r)
      A2[(size_t)(q0 + mb + r) * C + h * 64 + f * 16 + cl] = f2bf(o[f][r]);
}

// ---------- kernel 4: output projection GEMM + bias (f32 out) ----------
__global__ __launch_bounds__(256) void proj_gemm(
    const __bf16* __restrict__ A2, const __bf16* __restrict__ wp,
    const float* __restrict__ pb, float* __restrict__ out) {
  const int C = 1024;
  const int wid  = threadIdx.x >> 5;
  const int lane = threadIdx.x & 31;
  const int m0 = blockIdx.x * 64  + (wid & 1) * 32;
  const int n0 = blockIdx.y * 256 + (wid >> 1) * 64;

  v8f c[2][4];
#pragma unroll
  for (int i = 0; i < 2; ++i)
#pragma unroll
    for (int f = 0; f < 4; ++f) { v8f z = {}; c[i][f] = z; }

  for (int kk = 0; kk < C; kk += 32) {
    if (kk + 64 < C) {
      __builtin_prefetch(A2 + (size_t)m0 * C + kk + 64, 0, 1);
      __builtin_prefetch(wp + (size_t)n0 * C + kk + 64, 0, 1);
    }
    v16bf a0 = load_frag16(A2 + (size_t)m0 * C + kk, C);
    v16bf a1 = load_frag16(A2 + (size_t)(m0 + 16) * C + kk, C);
    v16bf b0 = load_frag16(wp + (size_t)(n0 +  0) * C + kk, C);
    v16bf b1 = load_frag16(wp + (size_t)(n0 + 16) * C + kk, C);
    v16bf b2 = load_frag16(wp + (size_t)(n0 + 32) * C + kk, C);
    v16bf b3 = load_frag16(wp + (size_t)(n0 + 48) * C + kk, C);
    c[0][0] = wmma_bf16(a0, b0, c[0][0]);
    c[0][1] = wmma_bf16(a0, b1, c[0][1]);
    c[0][2] = wmma_bf16(a0, b2, c[0][2]);
    c[0][3] = wmma_bf16(a0, b3, c[0][3]);
    c[1][0] = wmma_bf16(a1, b0, c[1][0]);
    c[1][1] = wmma_bf16(a1, b1, c[1][1]);
    c[1][2] = wmma_bf16(a1, b2, c[1][2]);
    c[1][3] = wmma_bf16(a1, b3, c[1][3]);
  }

  const int cl = lane & 15;
  const int mb = (lane >> 4) << 3;
#pragma unroll
  for (int i = 0; i < 2; ++i)
#pragma unroll
    for (int f = 0; f < 4; ++f) {
      int j = n0 + f * 16 + cl;
      float bias = pb[j];
#pragma unroll
      for (int r = 0; r < 8; ++r) {
        int m = m0 + i * 16 + mb + r;
        out[(size_t)m * C + j] = c[i][f][r] + bias;
      }
    }
}

// ---------- launch ----------

extern "C" void kernel_launch(void* const* d_in, const int* in_sizes, int n_in,
                              void* d_out, int out_size, void* d_ws, size_t ws_size,
                              hipStream_t stream) {
  const float* x      = (const float*)d_in[0];   // [4096,1024]
  const float* qkv_w  = (const float*)d_in[1];   // [3072,1024]
  const float* qkv_b  = (const float*)d_in[2];   // [3072]
  const float* proj_w = (const float*)d_in[3];   // [1024,1024]
  const float* proj_b = (const float*)d_in[4];   // [1024]
  float* out = (float*)d_out;

  char* ws = (char*)d_ws;
  __bf16* xb     = (__bf16*)(ws);                         //  8 MB [4096][1024]
  __bf16* wqkvb  = (__bf16*)(ws +  8388608);              //  6 MB [3072][1024]
  __bf16* wprojb = (__bf16*)(ws + 14680064);              //  2 MB [1024][1024]
  __bf16* Qs     = (__bf16*)(ws + 16777216);              //  8 MB [16][4096][64]
  __bf16* Ks     = (__bf16*)(ws + 25165824);              //  8 MB [16][4096][64]
  __bf16* Vt     = (__bf16*)(ws + 33554432);              //  8 MB [16][64][4096]
  __bf16* A2     = (__bf16*)(ws + 41943040);              //  8 MB [4096][1024]

  cvt_f32_bf16<<<4194304 / 256, 256, 0, stream>>>(x, xb, 4194304);
  cvt_f32_bf16<<<3145728 / 256, 256, 0, stream>>>(qkv_w, wqkvb, 3145728);
  cvt_f32_bf16<<<1048576 / 256, 256, 0, stream>>>(proj_w, wprojb, 1048576);

  qkv_gemm<<<dim3(64, 12), 256, 0, stream>>>(xb, wqkvb, qkv_b, Qs, Ks, Vt);
  attn_kernel<<<dim3(32, 16), 256, 0, stream>>>(Qs, Ks, Vt, A2);
  proj_gemm<<<dim3(64, 4), 256, 0, stream>>>(A2, wprojb, proj_b, out);
}
